// SAGELayer_352187318569
// MI455X (gfx1250) — matrix operand order repaired and checked by reading, not compiled, not run on hardware
//
#include <hip/hip_runtime.h>

#define NN 50000
#define NE 800000

typedef __attribute__((ext_vector_type(16))) __bf16 v16bf;
typedef __attribute__((ext_vector_type(8)))  float  v8f;

union Frag16 { unsigned u[8]; uint4 q[2]; v16bf v; };

// Split an fp32 pair into packed-bf16 hi dword and lo dword.
// hi = truncated bf16 (so x - hi is exact); lo = rounded bf16 of the residual.
__device__ __forceinline__ void bfsplit2(float x0, float x1, unsigned& hp, unsigned& lp) {
    unsigned b0 = __float_as_uint(x0), b1 = __float_as_uint(x1);
    unsigned h0 = b0 & 0xffff0000u, h1 = b1 & 0xffff0000u;
    hp = (h0 >> 16) | h1;
    unsigned r0 = __float_as_uint(x0 - __uint_as_float(h0)) + 0x8000u;
    unsigned r1 = __float_as_uint(x1 - __uint_as_float(h1)) + 0x8000u;
    lp = (r0 >> 16) | (r1 & 0xffff0000u);
}

__device__ __forceinline__ void bfsplit1(float x, unsigned short& h, unsigned short& l) {
    unsigned xb = __float_as_uint(x);
    unsigned hb = xb & 0xffff0000u;
    h = (unsigned short)(hb >> 16);
    unsigned rb = __float_as_uint(x - __uint_as_float(hb));
    l = (unsigned short)((rb + 0x8000u) >> 16);
}

// ---------------- zero scratch ----------------
__global__ void zero_kernel(float* __restrict__ p, int n) {
    int i = blockIdx.x * blockDim.x + threadIdx.x;
    if (i < n) p[i] = 0.0f;
}

// ---------------- weight prep: fp32 [out,in] -> bf16 hi/lo B-fragments ----------------
// B[k][n] = W[n][k].  CDNA5 16-bit B layout: VGPR v, h=lane/16: K=kc*32+8h+2(v&3)(+1)+(v>=4?16:0),
// N=lane%16.   Storage: ((kc*4+nt)*8+v)*32+lane
__global__ void prep_weights(const float* __restrict__ Wa, const float* __restrict__ We,
                             unsigned* __restrict__ WAh, unsigned* __restrict__ WAl,
                             unsigned* __restrict__ WEh, unsigned* __restrict__ WEl) {
    int id = blockIdx.x * 256 + threadIdx.x;
    if (id < 4096) {  // W_apply: 64 out x 128 in -> 4 kc x 4 nt
        int lane = id & 31, v = (id >> 5) & 7, nt = (id >> 8) & 3, kc = id >> 10;
        int k0 = kc * 32 + ((lane & 16) ? 8 : 0) + ((v & 4) ? 16 : 0) + 2 * (v & 3);
        int n  = nt * 16 + (lane & 15);
        unsigned short h0, l0, h1, l1;
        bfsplit1(Wa[n * 128 + k0],     h0, l0);
        bfsplit1(Wa[n * 128 + k0 + 1], h1, l1);
        WAh[id] = (unsigned)h0 | ((unsigned)h1 << 16);
        WAl[id] = (unsigned)l0 | ((unsigned)l1 << 16);
    } else if (id < 4096 + 6144) {  // edge_w: 64 out x 192 in -> 6 kc x 4 nt
        int id2 = id - 4096;
        int lane = id2 & 31, v = (id2 >> 5) & 7, nt = (id2 >> 8) & 3, kc = id2 >> 10;
        int k0 = kc * 32 + ((lane & 16) ? 8 : 0) + ((v & 4) ? 16 : 0) + 2 * (v & 3);
        int n  = nt * 16 + (lane & 15);
        unsigned short h0, l0, h1, l1;
        bfsplit1(We[n * 192 + k0],     h0, l0);
        bfsplit1(We[n * 192 + k0 + 1], h1, l1);
        WEh[id2] = (unsigned)h0 | ((unsigned)h1 << 16);
        WEl[id2] = (unsigned)l0 | ((unsigned)l1 << 16);
    }
}

// ---------------- segment sum + counts via f32 atomics ----------------
__global__ __launch_bounds__(256) void scatter_kernel(const float* __restrict__ ef,
                                                      const int* __restrict__ dst,
                                                      float* __restrict__ seg,
                                                      float* __restrict__ cnt) {
    long long id = (long long)blockIdx.x * 256 + threadIdx.x;  // NE*16 threads
    if (id >= (long long)NE * 16) return;
    int e = (int)(id >> 4), c = (int)(id & 15);
    int d = dst[e];
    float4 v = *(const float4*)(ef + (size_t)e * 64 + c * 4);
    float* p = seg + (size_t)d * 64 + c * 4;
    atomicAdd(p + 0, v.x);
    atomicAdd(p + 1, v.y);
    atomicAdd(p + 2, v.z);
    atomicAdd(p + 3, v.w);
    if (c == 0) atomicAdd(cnt + d, 1.0f);
}

// Wave-level 16x16 tile compute: acc += A(hi+lo) x B(hi+lo), bf16x3.
// A planes are pre-packed bf16 pairs in LDS; each fragment = 2 ds_load_b128.
template <int KC, int STRIDE>
__device__ __forceinline__ v8f tile_gemm(const unsigned* __restrict__ rowH,
                                         const unsigned* __restrict__ rowL,
                                         const unsigned* __restrict__ Bh,
                                         const unsigned* __restrict__ Bl,
                                         int nt, int lane, int kh4) {
    v8f acc = {};
#pragma unroll
    for (int kc = 0; kc < KC; kc++) {
        Frag16 ah, al, bh, bl;
        ah.q[0] = *(const uint4*)(rowH + kc * 16 + kh4);
        ah.q[1] = *(const uint4*)(rowH + kc * 16 + 8 + kh4);
        al.q[0] = *(const uint4*)(rowL + kc * 16 + kh4);
        al.q[1] = *(const uint4*)(rowL + kc * 16 + 8 + kh4);
#pragma unroll
        for (int v = 0; v < 8; v++) {
            int bi = ((kc * 4 + nt) * 8 + v) * 32 + lane;
            bh.u[v] = Bh[bi];
            bl.u[v] = Bl[bi];
        }
        acc = __builtin_amdgcn_wmma_f32_16x16x32_bf16(false, ah.v, false, bh.v, (short)0, acc, false, false);
        acc = __builtin_amdgcn_wmma_f32_16x16x32_bf16(false, al.v, false, bh.v, (short)0, acc, false, false);
        acc = __builtin_amdgcn_wmma_f32_16x16x32_bf16(false, ah.v, false, bl.v, (short)0, acc, false, false);
    }
    return acc;
}

// ---------------- node apply: h = relu([nfeats || seg/cnt] x Wa^T + ba) ----------------
__global__ __launch_bounds__(256) void node_apply_kernel(
    const float* __restrict__ nfeats, const float* __restrict__ seg,
    const float* __restrict__ cnt, const unsigned* __restrict__ Bh,
    const unsigned* __restrict__ Bl, const float* __restrict__ bias,
    float* __restrict__ hout) {
    // 32 rows x 64 packed pairs (K=128), stride 68 dwords (68%64==4: conflict-free, 16B aligned)
    __shared__ __align__(16) unsigned lh[32 * 68];
    __shared__ __align__(16) unsigned ll[32 * 68];
    int t = threadIdx.x;
    int nb = blockIdx.x * 32;

    for (int idx = t; idx < 2048; idx += 256) {  // 32 rows x 64 pairs
        int nl = idx >> 6, p = idx & 63;
        int node = nb + nl;
        float x0 = 0.f, x1 = 0.f;
        if (node < NN) {
            if (p < 32) {
                float2 v = *(const float2*)(nfeats + (size_t)node * 64 + p * 2);
                x0 = v.x; x1 = v.y;
            } else {
                float s = 1.0f / fmaxf(cnt[node], 1.0f);
                float2 v = *(const float2*)(seg + (size_t)node * 64 + (p - 32) * 2);
                x0 = v.x * s; x1 = v.y * s;
            }
        }
        unsigned hp, lp;
        bfsplit2(x0, x1, hp, lp);
        lh[nl * 68 + p] = hp;
        ll[nl * 68 + p] = lp;
    }
    __syncthreads();

    int lane = t & 31;
    int w = t >> 5;
    int mt = w >> 2, nt = w & 3;
    int row = mt * 16 + (lane & 15);
    int kh4 = (lane & 16) ? 4 : 0;

    v8f acc = tile_gemm<4, 68>(&lh[row * 68], &ll[row * 68], Bh, Bl, nt, lane, kh4);

    int col = nt * 16 + (lane & 15);
    float b = bias[col];
    int mbase = nb + mt * 16 + ((lane & 16) ? 8 : 0);
#pragma unroll
    for (int r = 0; r < 8; r++) {
        int node = mbase + r;
        if (node < NN) {
            float vv = acc[r] + b;
            hout[(size_t)node * 64 + col] = vv > 0.f ? vv : 0.f;
        }
    }
}

// ---------------- edge update: eh = relu([h_src || h_dst || e] x We^T + be) ----------------
__global__ __launch_bounds__(256) void edge_update_kernel(
    const float* __restrict__ h, const float* __restrict__ ef,
    const int* __restrict__ src, const int* __restrict__ dst,
    const unsigned* __restrict__ Bh, const unsigned* __restrict__ Bl,
    const float* __restrict__ bias, float* __restrict__ eh) {
    // 32 rows x 96 packed pairs (K=192), stride 100 dwords (100%64==36: conflict-free, 16B aligned)
    __shared__ __align__(16) unsigned lh[32 * 100];
    __shared__ __align__(16) unsigned ll[32 * 100];
    __shared__ int sIdx[32], dIdx[32];
    int t = threadIdx.x;
    int e0 = blockIdx.x * 32;
    if (t < 32) sIdx[t] = src[e0 + t];
    else if (t < 64) dIdx[t - 32] = dst[e0 + t - 32];
    __syncthreads();

    for (int idx = t; idx < 3072; idx += 256) {  // 32 rows x 96 pairs
        int el = idx / 96, p = idx - el * 96;
        int sec = p >> 5, k2 = p & 31;
        const float* base;
        if (sec == 0)      base = h  + (size_t)sIdx[el] * 64;
        else if (sec == 1) base = h  + (size_t)dIdx[el] * 64;
        else               base = ef + (size_t)(e0 + el) * 64;
        float2 v = *(const float2*)(base + k2 * 2);
        unsigned hp, lp;
        bfsplit2(v.x, v.y, hp, lp);
        lh[el * 100 + p] = hp;
        ll[el * 100 + p] = lp;
    }
    __syncthreads();

    int lane = t & 31;
    int w = t >> 5;
    int mt = w >> 2, nt = w & 3;
    int row = mt * 16 + (lane & 15);
    int kh4 = (lane & 16) ? 4 : 0;

    v8f acc = tile_gemm<6, 100>(&lh[row * 100], &ll[row * 100], Bh, Bl, nt, lane, kh4);

    int col = nt * 16 + (lane & 15);
    float b = bias[col];
    int ebase = e0 + mt * 16 + ((lane & 16) ? 8 : 0);
#pragma unroll
    for (int r = 0; r < 8; r++) {
        float vv = acc[r] + b;
        eh[(size_t)(ebase + r) * 64 + col] = vv > 0.f ? vv : 0.f;
    }
}

extern "C" void kernel_launch(void* const* d_in, const int* in_sizes, int n_in,
                              void* d_out, int out_size, void* d_ws, size_t ws_size,
                              hipStream_t stream) {
    const float* nfeats = (const float*)d_in[0];
    const float* efeats = (const float*)d_in[1];
    const int*   src    = (const int*)d_in[2];
    const int*   dst    = (const int*)d_in[3];
    const float* Wa     = (const float*)d_in[4];
    const float* ba     = (const float*)d_in[5];
    const float* We     = (const float*)d_in[6];
    const float* be     = (const float*)d_in[7];
    (void)in_sizes; (void)n_in; (void)out_size; (void)ws_size;

    float* hout  = (float*)d_out;
    float* ehout = hout + (size_t)NN * 64;

    // workspace carve (all 16B aligned): seg 12.8MB | cnt 200KB | B-fragment tables
    float*    seg = (float*)d_ws;
    float*    cnt = seg + (size_t)NN * 64;
    unsigned* WAh = (unsigned*)(cnt + NN);
    unsigned* WAl = WAh + 4096;
    unsigned* WEh = WAl + 4096;
    unsigned* WEl = WEh + 6144;

    int nz = NN * 64 + NN;
    zero_kernel<<<(nz + 255) / 256, 256, 0, stream>>>(seg, nz);
    prep_weights<<<(4096 + 6144 + 255) / 256, 256, 0, stream>>>(Wa, We, WAh, WAl, WEh, WEl);
    scatter_kernel<<<(NE * 16) / 256, 256, 0, stream>>>(efeats, dst, seg, cnt);
    node_apply_kernel<<<(NN + 31) / 32, 256, 0, stream>>>(nfeats, seg, cnt, WAh, WAl, ba, hout);
    edge_update_kernel<<<NE / 32, 256, 0, stream>>>(hout, efeats, src, dst, WEh, WEl, be, ehout);
}
